// LSTMCell_29987461661321
// MI455X (gfx1250) — matrix operand order repaired and checked by reading, not compiled, not run on hardware
//
#include <hip/hip_runtime.h>
#include <hip/hip_bf16.h>
#include <math.h>

// ---------------------------------------------------------------------------
// Types for CDNA5 WMMA (wave32, v_wmma_f32_16x16x32_bf16)
// ---------------------------------------------------------------------------
typedef __attribute__((ext_vector_type(8)))  float          v8f;
typedef __attribute__((ext_vector_type(16))) __bf16         v16bf;
typedef __attribute__((ext_vector_type(16))) unsigned short v16u;
typedef __attribute__((ext_vector_type(8)))  unsigned short v8u;

#define HW   262144      // 512*512
#define EPSF 1e-5f
#define R2HW 268324      // 518*518

// Optional CDNA5 async global->LDS path (guarded; falls back to uint4 copies)
#if defined(__has_builtin)
#if __has_builtin(__builtin_amdgcn_global_load_async_to_lds_b128) && \
    __has_builtin(__builtin_amdgcn_s_wait_asynccnt)
#define USE_ASYNC_LDS 1
#endif
#endif

#ifdef USE_ASYNC_LDS
typedef __attribute__((__vector_size__(4 * sizeof(int)))) int vi4_t;
__device__ __forceinline__ void async_copy16(const void* g, void* l) {
    __builtin_amdgcn_global_load_async_to_lds_b128((vi4_t*)g, (vi4_t*)l, 0, 0);
}
__device__ __forceinline__ void async_wait0() {
    __builtin_amdgcn_s_wait_asynccnt(0);
}
#endif

__device__ __forceinline__ unsigned short f2bf(float f) {
    unsigned int u = __float_as_uint(f);
    unsigned int r = u + 0x7FFFu + ((u >> 16) & 1u);   // round-to-nearest-even
    return (unsigned short)(r >> 16);
}
__device__ __forceinline__ float bf2f(unsigned short h) {
    return __uint_as_float(((unsigned int)h) << 16);
}
__device__ __forceinline__ float sigmoidf_(float x) { return 1.f / (1.f + __expf(-x)); }

// ---------------------------------------------------------------------------
// zero the stats region (capture-safe replacement for memset)
// ---------------------------------------------------------------------------
__global__ void zero_stats_kernel(float* stats) {
    int i = threadIdx.x;
    if (i < 512) stats[i] = 0.f;
}

// ---------------------------------------------------------------------------
// weight pre-pack: f32 [32][CIN][3][3]  ->  bf16 [32][K], K = (ky*3+kx)*CIN+cin
// ---------------------------------------------------------------------------
template <int CIN>
__global__ __launch_bounds__(256) void pack_weights_kernel(
    const float* __restrict__ w, unsigned short* __restrict__ wpk)
{
    constexpr int K = CIN * 9;
    int idx = blockIdx.x * 256 + threadIdx.x;
    if (idx >= 32 * K) return;
    int m = idx / K, k = idx - m * K;
    int p = k / CIN, cin = k - p * CIN;   // p = ky*3+kx
    wpk[idx] = f2bf(w[(m * CIN + cin) * 9 + p]);
}

// ---------------------------------------------------------------------------
// e1: direct 3x3 conv 4->32, pad 1, f32 NCHW in/out, per-channel stats
// ---------------------------------------------------------------------------
__global__ __launch_bounds__(256) void conv_e1_kernel(
    const float* __restrict__ x, const float* __restrict__ w,
    float* __restrict__ out, float* __restrict__ stats)
{
    __shared__ float s_w[1152];          // 32*4*9
    __shared__ float s_sum[32], s_sq[32];
    const int tid = threadIdx.x;
    for (int idx = tid; idx < 1152; idx += 256) s_w[idx] = w[idx];
    if (tid < 32) { s_sum[tid] = 0.f; s_sq[tid] = 0.f; }
    __syncthreads();

    const int pix = blockIdx.x * 256 + tid;
    const int py = pix >> 9, px = pix & 511;

    float patch[36];
#pragma unroll
    for (int cin = 0; cin < 4; ++cin)
#pragma unroll
        for (int ky = 0; ky < 3; ++ky)
#pragma unroll
            for (int kx = 0; kx < 3; ++kx) {
                int gy = py + ky - 1, gx = px + kx - 1;
                patch[cin * 9 + ky * 3 + kx] =
                    ((unsigned)gy < 512u && (unsigned)gx < 512u)
                        ? x[cin * HW + (gy << 9) + gx] : 0.f;
            }
    for (int m = 0; m < 32; ++m) {
        float acc = 0.f;
#pragma unroll
        for (int t = 0; t < 36; ++t) acc = fmaf(patch[t], s_w[m * 36 + t], acc);
        out[m * HW + pix] = acc;
        atomicAdd(&s_sum[m], acc);
        atomicAdd(&s_sq[m], acc * acc);
    }
    __syncthreads();
    if (tid < 32) {
        atomicAdd(&stats[tid],      s_sum[tid]);
        atomicAdd(&stats[32 + tid], s_sq[tid]);
    }
}

// ---------------------------------------------------------------------------
// BN (+optional ReLU): f32 NCHW pre-act -> bf16 NHWC (32 ch packed per pixel)
// outb already includes the channel offset; OSTRIDE = channels of dest buffer.
// ---------------------------------------------------------------------------
template <int RELU, int OSTRIDE>
__global__ __launch_bounds__(256) void bn_act_kernel(
    const float* __restrict__ pre, const float* __restrict__ stats,
    unsigned short* __restrict__ outb, float invN)
{
    __shared__ float s_mean[32], s_scale[32];
    const int tid = threadIdx.x;
    if (tid < 32) {
        float m = stats[tid] * invN;
        float var = stats[32 + tid] * invN - m * m;
        s_mean[tid] = m;
        s_scale[tid] = rsqrtf(var + EPSF);
    }
    __syncthreads();

    const int pix = blockIdx.x * 256 + tid;
    unsigned int pk[16];
#pragma unroll
    for (int ch = 0; ch < 32; ch += 2) {
        float v0 = (pre[ch * HW + pix] - s_mean[ch]) * s_scale[ch];
        float v1 = (pre[(ch + 1) * HW + pix] - s_mean[ch + 1]) * s_scale[ch + 1];
        if (RELU) { v0 = fmaxf(v0, 0.f); v1 = fmaxf(v1, 0.f); }
        pk[ch >> 1] = (unsigned)f2bf(v0) | ((unsigned)f2bf(v1) << 16);
    }
    uint4* dst = (uint4*)(outb + (size_t)pix * OSTRIDE);
#pragma unroll
    for (int q = 0; q < 4; ++q) {
        uint4 v; v.x = pk[4*q]; v.y = pk[4*q+1]; v.z = pk[4*q+2]; v.w = pk[4*q+3];
        dst[q] = v;
    }
}

// prev_h (f32 NCHW) -> bf16 NHWC into channels 0..31 of the 64-ch xh buffer
__global__ __launch_bounds__(256) void pack_prevh_kernel(
    const float* __restrict__ in, unsigned short* __restrict__ xh)
{
    const int pix = blockIdx.x * 256 + threadIdx.x;
    unsigned int pk[16];
#pragma unroll
    for (int ch = 0; ch < 32; ch += 2) {
        pk[ch >> 1] = (unsigned)f2bf(in[ch * HW + pix]) |
                      ((unsigned)f2bf(in[(ch + 1) * HW + pix]) << 16);
    }
    uint4* dst = (uint4*)(xh + (size_t)pix * 64);
#pragma unroll
    for (int q = 0; q < 4; ++q) {
        uint4 v; v.x = pk[4*q]; v.y = pk[4*q+1]; v.z = pk[4*q+2]; v.w = pk[4*q+3];
        dst[q] = v;
    }
}

// ---------------------------------------------------------------------------
// Implicit-GEMM 3x3 conv (CIN in {32,64} -> 32), v_wmma_f32_16x16x32_bf16.
//  - activations NHWC bf16  -> haloed LDS tile [6 rows][130 x][CIN] is a set
//    of byte-contiguous row copies (async global->LDS capable)
//  - weights pre-packed bf16 [32][K] -> contiguous LDS memcpy
//  - loop kk outer / row inner: A fragments reused across 4 output rows
//  - block = 256 thr (8 waves), tile = 128 px * 4 rows * 32 out channels
// ---------------------------------------------------------------------------
template <int CIN>
__global__ __launch_bounds__(256) void conv3x3_wmma_kernel(
    const unsigned short* __restrict__ in,    // NHWC [512][512][CIN] bf16
    const unsigned short* __restrict__ wpk,   // [32][K] bf16 pre-packed
    float* __restrict__ out,                  // NCHW [32][512][512] pre-BN f32
    float* __restrict__ stats)                // [2][32]
{
    constexpr int K = CIN * 9;
    constexpr int ROWEL = 130 * CIN;          // LDS elements per haloed row
    extern __shared__ __align__(32) unsigned char smem[];
    unsigned short* s_in = (unsigned short*)smem;        // [6][130][CIN]
    unsigned short* s_w  = s_in + 6 * ROWEL;             // [32][K]
    float* s_sum = (float*)(s_w + 32 * K);               // 32
    float* s_sq  = s_sum + 32;                           // 32

    const int tid   = threadIdx.x;
    const int x0    = blockIdx.x * 128;
    const int ybase = blockIdx.y * 4;

    // ---- stage 6 haloed input rows (contiguous 16B chunks) ----
    for (int e = tid * 8; e < 6 * ROWEL; e += 2048) {
        int r   = e / ROWEL;
        int rem = e - r * ROWEL;
        int xx  = rem / CIN;                  // 0..129
        int gy = ybase - 1 + r, gx = x0 - 1 + xx;
        bool ok = ((unsigned)gy < 512u) && ((unsigned)gx < 512u);
        if (ok) {
            const unsigned short* src =
                in + ((size_t)(gy << 9) + gx) * CIN + (rem - xx * CIN);
#ifdef USE_ASYNC_LDS
            async_copy16(src, s_in + e);
#else
            *(uint4*)(s_in + e) = *(const uint4*)src;
#endif
        } else {
            uint4 z; z.x = 0; z.y = 0; z.z = 0; z.w = 0;
            *(uint4*)(s_in + e) = z;
        }
    }
    // ---- stage pre-packed weights (pure contiguous memcpy) ----
    for (int e = tid * 8; e < 32 * K; e += 2048) {
#ifdef USE_ASYNC_LDS
        async_copy16(wpk + e, s_w + e);
#else
        *(uint4*)(s_w + e) = *(const uint4*)(wpk + e);
#endif
    }
    if (tid < 64) s_sum[tid] = 0.f;           // covers s_sum + s_sq
#ifdef USE_ASYNC_LDS
    async_wait0();
#endif
    __syncthreads();

    const int lane  = tid & 31;
    const int n     = lane & 15;     // B column / A row (M)
    const int group = lane >> 4;
    const int wv    = tid >> 5;      // wave id 0..7
    const int px    = wv * 16;       // pixel offset inside 128-px tile

    v8f zero8 = {};
    v8f acc0[4], acc1[4];
#pragma unroll
    for (int r = 0; r < 4; ++r) { acc0[r] = zero8; acc1[r] = zero8; }

    const unsigned short* swn0 = s_w + n * K;
    const unsigned short* swn1 = s_w + (n + 16) * K;

#pragma unroll
    for (int kk = 0; kk < K / 32; ++kk) {
        const int kbase = kk * 32;
        const int p    = kbase / CIN;         // tap index (ky*3+kx)
        const int cin0 = kbase - p * CIN;
        const int ky = p / 3, kx = p - ky * 3;

        // A fragments for both M tiles: K = g*16+h contiguous
        v16u au0 = *(const v16u*)(swn0 + kbase + group * 16);
        v16u au1 = *(const v16u*)(swn1 + kbase + group * 16);
        v16bf a0 = __builtin_bit_cast(v16bf, au0);
        v16bf a1 = __builtin_bit_cast(v16bf, au1);

#pragma unroll
        for (int ry = 0; ry < 4; ++ry) {
            // B fragment: K = group*16 + h contiguous in cin (32B aligned)
            const unsigned short* bp =
                s_in + ((ry + ky) * 130 + px + n + kx) * CIN + cin0 + group * 16;
            v16u bu = *(const v16u*)bp;
            v16bf b = __builtin_bit_cast(v16bf, bu);
            acc0[ry] = __builtin_amdgcn_wmma_f32_16x16x32_bf16(
                           false, a0, false, b, (short)0, acc0[ry], false, false);
            acc1[ry] = __builtin_amdgcn_wmma_f32_16x16x32_bf16(
                           false, a1, false, b, (short)0, acc1[ry], false, false);
        }
    }

    // ---- store (NCHW) + per-channel statistics ----
    float as0[8] = {0,0,0,0,0,0,0,0}, aq0[8] = {0,0,0,0,0,0,0,0};
    float as1[8] = {0,0,0,0,0,0,0,0}, aq1[8] = {0,0,0,0,0,0,0,0};
    const int xo  = x0 + px + n;
    const int chb = group * 8;
#pragma unroll
    for (int ry = 0; ry < 4; ++ry) {
        float* o = out + ((ybase + ry) << 9) + xo;
#pragma unroll
        for (int v = 0; v < 8; ++v) {
            float v0 = acc0[ry][v], v1 = acc1[ry][v];
            o[(chb + v) * HW]      = v0;
            o[(chb + v + 16) * HW] = v1;
            as0[v] += v0; aq0[v] += v0 * v0;
            as1[v] += v1; aq1[v] += v1 * v1;
        }
    }
#pragma unroll
    for (int v = 0; v < 8; ++v) {
        atomicAdd(&s_sum[chb + v],      as0[v]);
        atomicAdd(&s_sq [chb + v],      aq0[v]);
        atomicAdd(&s_sum[chb + 16 + v], as1[v]);
        atomicAdd(&s_sq [chb + 16 + v], aq1[v]);
    }
    __syncthreads();
    if (tid < 32) {
        atomicAdd(&stats[tid],      s_sum[tid]);
        atomicAdd(&stats[32 + tid], s_sq[tid]);
    }
}

// ---------------------------------------------------------------------------
// LSTM pointwise (per-pixel): BN each gate inline, nonlinearities, update.
// Gate pre-acts are NCHW f32; next_h also emitted bf16 NHWC (r1 input).
// ---------------------------------------------------------------------------
__global__ __launch_bounds__(256) void lstm_pointwise_kernel(
    const float* __restrict__ pf, const float* __restrict__ pi,
    const float* __restrict__ pc, const float* __restrict__ po,
    const float* __restrict__ st,          // 4 stages x 64 floats
    const float* __restrict__ prev_c,
    float* __restrict__ out_c, float* __restrict__ out_h,
    unsigned short* __restrict__ h_bf)
{
    __shared__ float s_m[4][32], s_s[4][32];
    const int tid = threadIdx.x;
    if (tid < 128) {
        int gate = tid >> 5, ch = tid & 31;
        const float invN = 1.f / (float)HW;
        float m = st[gate * 64 + ch] * invN;
        float var = st[gate * 64 + 32 + ch] * invN - m * m;
        s_m[gate][ch] = m;
        s_s[gate][ch] = rsqrtf(var + EPSF);
    }
    __syncthreads();

    const int pix = blockIdx.x * 256 + tid;
    unsigned int pk[16];
#pragma unroll
    for (int ch = 0; ch < 32; ++ch) {
        int i = ch * HW + pix;
        float f  = sigmoidf_((pf[i] - s_m[0][ch]) * s_s[0][ch]);
        float ig = sigmoidf_((pi[i] - s_m[1][ch]) * s_s[1][ch]);
        float cg = tanhf(    (pc[i] - s_m[2][ch]) * s_s[2][ch]);
        float o  = sigmoidf_((po[i] - s_m[3][ch]) * s_s[3][ch]);
        float nc = prev_c[i] * f + ig * cg;
        float nh = tanhf(nc) * o;
        out_c[i] = nc;
        out_h[i] = nh;
        unsigned hb = f2bf(nh);
        if (ch & 1) pk[ch >> 1] |= hb << 16; else pk[ch >> 1] = hb;
    }
    uint4* dst = (uint4*)(h_bf + (size_t)pix * 32);
#pragma unroll
    for (int q = 0; q < 4; ++q) {
        uint4 v; v.x = pk[4*q]; v.y = pk[4*q+1]; v.z = pk[4*q+2]; v.w = pk[4*q+3];
        dst[q] = v;
    }
}

// ---------------------------------------------------------------------------
// r2: direct 3x3 conv 32->3, pad=4 -> 518x518 pre-BN + stats. NHWC bf16 in.
// ---------------------------------------------------------------------------
__global__ __launch_bounds__(256) void conv_r2_kernel(
    const unsigned short* __restrict__ in, const float* __restrict__ w,
    float* __restrict__ out, float* __restrict__ stats)
{
    __shared__ float s_w[864];           // 3*32*9
    __shared__ float s_sum[3], s_sq[3];
    const int tid = threadIdx.x;
    for (int idx = tid; idx < 864; idx += 256) s_w[idx] = w[idx];
    if (tid < 3) { s_sum[tid] = 0.f; s_sq[tid] = 0.f; }
    __syncthreads();

    const int pix = blockIdx.x * 256 + tid;
    if (pix < R2HW) {
        const int i = pix / 518, j = pix - i * 518;
        float acc[3] = {0.f, 0.f, 0.f};
#pragma unroll
        for (int ky = 0; ky < 3; ++ky)
#pragma unroll
            for (int kx = 0; kx < 3; ++kx) {
                int gy = i + ky - 4, gx = j + kx - 4;
                if ((unsigned)gy < 512u && (unsigned)gx < 512u) {
                    const int t = ky * 3 + kx;
                    const uint4* q =
                        (const uint4*)(in + ((size_t)(gy << 9) + gx) * 32);
                    unsigned short t16[32];
                    *(uint4*)(t16 +  0) = q[0];
                    *(uint4*)(t16 +  8) = q[1];
                    *(uint4*)(t16 + 16) = q[2];
                    *(uint4*)(t16 + 24) = q[3];
#pragma unroll
                    for (int cin = 0; cin < 32; ++cin) {
                        float xv = bf2f(t16[cin]);
#pragma unroll
                        for (int m = 0; m < 3; ++m)
                            acc[m] = fmaf(xv, s_w[m * 288 + cin * 9 + t], acc[m]);
                    }
                }
            }
#pragma unroll
        for (int m = 0; m < 3; ++m) {
            out[m * R2HW + pix] = acc[m];
            atomicAdd(&s_sum[m], acc[m]);
            atomicAdd(&s_sq[m], acc[m] * acc[m]);
        }
    }
    __syncthreads();
    if (tid < 3) {
        atomicAdd(&stats[tid],      s_sum[tid]);
        atomicAdd(&stats[32 + tid], s_sq[tid]);
    }
}

// ---------------------------------------------------------------------------
// Head: K=1 -> single-pixel gather at (hx+3, hy+3), BN, scalar weight + bias.
// ---------------------------------------------------------------------------
__global__ __launch_bounds__(256) void res_kernel(
    const int* __restrict__ holes, const float* __restrict__ r2pre,
    const float* __restrict__ stats,
    const float* __restrict__ w_oil, const float* __restrict__ b_oil,
    const float* __restrict__ w_wat, const float* __restrict__ b_wat,
    const float* __restrict__ w_gas, const float* __restrict__ b_gas,
    float* __restrict__ res)
{
    int nIdx = threadIdx.x;
    if (nIdx >= 256) return;
    int hx = holes[2 * nIdx], hy = holes[2 * nIdx + 1];
    const float invN = 1.f / (float)R2HW;
    float wq[3] = {w_oil[0], w_wat[0], w_gas[0]};
    float bq[3] = {b_oil[0], b_wat[0], b_gas[0]};
#pragma unroll
    for (int c = 0; c < 3; ++c) {
        float mean = stats[c] * invN;
        float var  = stats[32 + c] * invN - mean * mean;
        float v = (r2pre[c * R2HW + (hx + 3) * 518 + (hy + 3)] - mean)
                  * rsqrtf(var + EPSF);
        res[nIdx * 3 + c] = v * wq[c] + bq[c];
    }
}

// ---------------------------------------------------------------------------
// Host launcher
// ---------------------------------------------------------------------------
extern "C" void kernel_launch(void* const* d_in, const int* in_sizes, int n_in,
                              void* d_out, int out_size, void* d_ws, size_t ws_size,
                              hipStream_t stream)
{
    const float* x      = (const float*)d_in[0];
    const float* prev_c = (const float*)d_in[1];
    const float* prev_h = (const float*)d_in[2];
    const int*   holes  = (const int*)  d_in[3];
    const float* w_e1   = (const float*)d_in[4];
    const float* w_e2   = (const float*)d_in[5];
    const float* w_f    = (const float*)d_in[6];
    const float* w_i    = (const float*)d_in[7];
    const float* w_c    = (const float*)d_in[8];
    const float* w_o    = (const float*)d_in[9];
    const float* w_r1   = (const float*)d_in[10];
    const float* w_r2   = (const float*)d_in[11];
    const float* w_oil  = (const float*)d_in[12];
    const float* b_oil  = (const float*)d_in[13];
    const float* w_wat  = (const float*)d_in[14];
    const float* b_wat  = (const float*)d_in[15];
    const float* w_gas  = (const float*)d_in[16];
    const float* b_gas  = (const float*)d_in[17];

    float* out_c = (float*)d_out;
    float* out_h = out_c + 32 * HW;
    float* res   = out_c + 64 * HW;

    // workspace layout (~220 MB)
    char* ws = (char*)d_ws;
    float* stats = (float*)ws;                          // 8 stages x 64 f32
    size_t off = 4096;
    unsigned short* wpk = (unsigned short*)(ws + off); off += 40960;   // <=36864B
    float* buf_pre = (float*)(ws + off); off += (size_t)32 * HW * 4;
    float* gate_pre[4];
    for (int g = 0; g < 4; ++g) { gate_pre[g] = (float*)(ws + off); off += (size_t)32 * HW * 4; }
    float* r2_pre = (float*)(ws + off); off += (size_t)3 * R2HW * 4 + 256;
    unsigned short* a_bf  = (unsigned short*)(ws + off); off += (size_t)32 * HW * 2;  // NHWC 32ch
    unsigned short* xh_bf = (unsigned short*)(ws + off); off += (size_t)64 * HW * 2;  // NHWC 64ch
    (void)ws_size; (void)n_in; (void)in_sizes; (void)out_size;

    const float invN = 1.f / (float)HW;
    const int NPB = HW / 256;                           // 1024 pixel blocks
    const size_t SMEM32 = 6u*130*32*2 + 32u*288*2 + 256;   //  68608 B
    const size_t SMEM64 = 6u*130*64*2 + 32u*576*2 + 256;   // 136960 B
    const dim3 gconv(4, 128);

    zero_stats_kernel<<<1, 512, 0, stream>>>(stats);

    // e1 -> BN+ReLU -> bf16 NHWC
    conv_e1_kernel<<<1024, 256, 0, stream>>>(x, w_e1, buf_pre, stats + 0);
    bn_act_kernel<1, 32><<<NPB, 256, 0, stream>>>(buf_pre, stats + 0, a_bf, invN);

    // e2 (WMMA) -> BN -> x_emb into channels 32..63 of xh; prev_h -> ch 0..31
    pack_weights_kernel<32><<<36, 256, 0, stream>>>(w_e2, wpk);
    conv3x3_wmma_kernel<32><<<gconv, 256, SMEM32, stream>>>(a_bf, wpk, buf_pre, stats + 64);
    bn_act_kernel<0, 64><<<NPB, 256, 0, stream>>>(buf_pre, stats + 64, xh_bf + 32, invN);
    pack_prevh_kernel<<<NPB, 256, 0, stream>>>(prev_h, xh_bf);

    // four gate convs 64->32 (WMMA)
    const float* wg[4] = {w_f, w_i, w_c, w_o};
    for (int g = 0; g < 4; ++g) {
        pack_weights_kernel<64><<<72, 256, 0, stream>>>(wg[g], wpk);
        conv3x3_wmma_kernel<64><<<gconv, 256, SMEM64, stream>>>(
            xh_bf, wpk, gate_pre[g], stats + (2 + g) * 64);
    }

    // pointwise LSTM update; next_h also stored bf16 NHWC (r1 input)
    lstm_pointwise_kernel<<<NPB, 256, 0, stream>>>(
        gate_pre[0], gate_pre[1], gate_pre[2], gate_pre[3],
        stats + 2 * 64, prev_c, out_c, out_h, a_bf);

    // r1 (WMMA) -> BN+ReLU -> bf16 NHWC (reuse xh buffer, 32ch stride)
    pack_weights_kernel<32><<<36, 256, 0, stream>>>(w_r1, wpk);
    conv3x3_wmma_kernel<32><<<gconv, 256, SMEM32, stream>>>(a_bf, wpk, buf_pre, stats + 6 * 64);
    bn_act_kernel<1, 32><<<NPB, 256, 0, stream>>>(buf_pre, stats + 6 * 64,
                                                  (unsigned short*)xh_bf, invN);

    // r2 (pad=4, 518x518) + head
    conv_r2_kernel<<<(R2HW + 255) / 256, 256, 0, stream>>>(
        (unsigned short*)xh_bf, w_r2, r2_pre, stats + 7 * 64);
    res_kernel<<<1, 256, 0, stream>>>(holes, r2_pre, stats + 7 * 64,
                                      w_oil, b_oil, w_wat, b_wat, w_gas, b_gas, res);
}
